// Generator_25022479466570
// MI455X (gfx1250) — compile-verified, matrix operand order checked by reading, not run on hardware
//
#include <hip/hip_runtime.h>
#include <cstddef>

// ---------------------------------------------------------------------------
// Types
// ---------------------------------------------------------------------------
typedef __bf16 v16bf __attribute__((ext_vector_type(16)));
typedef __bf16 v8bf  __attribute__((ext_vector_type(8)));
typedef float  v8f   __attribute__((ext_vector_type(8)));
typedef unsigned int u32x4 __attribute__((ext_vector_type(4)));
typedef int          i32x8 __attribute__((ext_vector_type(8)));

#define Bsz 64
#define Tn  512
#define Ln  1024
#define ND  512        // NOTE_DIM
#define H4  2048       // 4*ND
#define EMB 32
#define LOUT 1024

// ---------------------------------------------------------------------------
// Workspace layout (bytes, all 256-aligned)
// ---------------------------------------------------------------------------
static constexpr size_t OFF_EMB = 0;                              // (B,T,32) bf16
static constexpr size_t OFF_WI1 = OFF_EMB + (size_t)Bsz*Tn*EMB*2; // 2048x32  bf16 blob
static constexpr size_t OFF_WH1 = OFF_WI1 + (size_t)H4*EMB*2;     // 2048x512 bf16 blob
static constexpr size_t OFF_WI2 = OFF_WH1 + (size_t)H4*ND*2;
static constexpr size_t OFF_WH2 = OFF_WI2 + (size_t)H4*ND*2;
static constexpr size_t OFF_W2  = OFF_WH2 + (size_t)H4*ND*2;      // 512x512 bf16 blob
static constexpr size_t OFF_FB  = OFF_W2  + (size_t)ND*ND*2;      // (B,512) f32
static constexpr size_t OFF_C1  = OFF_FB  + (size_t)Bsz*ND*4;     // (B,T,512) bf16
static constexpr size_t OFF_C2  = OFF_C1  + (size_t)Bsz*Tn*ND*2;  // (B,T,512) bf16
static constexpr size_t OFF_ALL = OFF_C2  + (size_t)Bsz*Tn*ND*2;  // (B,T,512) f32

// ---------------------------------------------------------------------------
// Helpers
// ---------------------------------------------------------------------------
__device__ __forceinline__ unsigned short f2bf(float x) {
  unsigned int u = __float_as_uint(x);
  unsigned int r = u + 0x7fffu + ((u >> 16) & 1u);
  return (unsigned short)(r >> 16);
}
__device__ __forceinline__ float sigm(float x) { return 1.f / (1.f + __expf(-x)); }

__device__ __forceinline__ v8f zero8() {
  v8f z;
#pragma unroll
  for (int i = 0; i < 8; ++i) z[i] = 0.f;
  return z;
}

__device__ __forceinline__ v8f wmma_bf16(v16bf a, v16bf b, v8f c) {
  return __builtin_amdgcn_wmma_f32_16x16x32_bf16(false, a, false, b, (short)0, c, false, false);
}

// A fragment (16x32 bf16) from a row-major matrix (row = M, contiguous K).
// Lane layout per ISA 7.12.2: lanes 0-15 hold M=lane, hi half-K split by lane>>4.
__device__ __forceinline__ v16bf loadA(const unsigned short* base, long rowStride, int kTile) {
  const int lane = threadIdx.x & 31;
  const int m = lane & 15, hi = lane >> 4;
  const unsigned short* p = base + (long)m * rowStride + kTile * 32 + hi * 8;
  v8bf a0 = *(const v8bf*)p;
  v8bf a1 = *(const v8bf*)(p + 16);
  v16bf r;
#pragma unroll
  for (int i = 0; i < 8; ++i) { r[i] = a0[i]; r[i + 8] = a1[i]; }
  return r;
}

// B fragment (32x16 bf16) from a pre-packed tile blob: tile (nTile,kTile) is 512
// contiguous elements, lane's 16 elements contiguous (32B per lane).
__device__ __forceinline__ v16bf loadB(const unsigned short* blob, int nTile, int kTile, int kTiles) {
  const int lane = threadIdx.x & 31;
  const unsigned short* p = blob + ((size_t)(nTile * kTiles + kTile)) * 512 + lane * 16;
  return *(const v16bf*)p;
}

// ---------------------------------------------------------------------------
// Kernel: pack fp32 (N x ldsrc) weight into bf16 WMMA-B tile blob
// ---------------------------------------------------------------------------
__global__ void pack_b_kernel(const float* __restrict__ src, int ldsrc, int koff,
                              unsigned short* __restrict__ dst, int N, int K) {
  const int e = blockIdx.x * 256 + threadIdx.x;
  if (e >= N * K) return;
  const int kTiles = K / 32;
  const int tile = e >> 9, within = e & 511;
  const int lane = within >> 4, j = within & 15;
  const int nTile = tile / kTiles, kTile = tile % kTiles;
  const int hi = lane >> 4;
  const int k = kTile * 32 + ((j < 8) ? (hi * 8 + j) : (16 + hi * 8 + (j - 8)));
  const int n = nTile * 16 + (lane & 15);
  dst[e] = f2bf(src[(size_t)n * ldsrc + koff + k]);
}

// ---------------------------------------------------------------------------
// Kernel: embedding gather -> bf16 (B*T, 32)
// ---------------------------------------------------------------------------
__global__ void embed_kernel(const int* __restrict__ trees, const float* __restrict__ table,
                             unsigned short* __restrict__ emb) {
  const int e = blockIdx.x * 256 + threadIdx.x;
  if (e >= Bsz * Tn * EMB) return;
  const int k = e & 31, bt = e >> 5;
  const int tok = trees[(size_t)bt * 3 + 2];
  emb[e] = f2bf(table[(size_t)tok * EMB + k]);
}

// ---------------------------------------------------------------------------
// Kernel: fb[b][n] = first[b] . (nextW[n,:512]+nextW[n,512:1024]) + nextb[n]
// ---------------------------------------------------------------------------
__global__ void fb_kernel(const float* __restrict__ first, const float* __restrict__ nextW,
                          const float* __restrict__ nextb, float* __restrict__ fb) {
  const int e = blockIdx.x * 256 + threadIdx.x;
  if (e >= Bsz * ND) return;
  const int b = e >> 9, n = e & 511;
  const float* fr = first + (size_t)b * ND;
  const float* wr = nextW + (size_t)n * (3 * ND);
  float s = nextb[n];
  for (int k = 0; k < ND; ++k) s += fr[k] * (wr[k] + wr[ND + k]);
  fb[e] = s;
}

// ---------------------------------------------------------------------------
// Kernel: persistent single LSTM layer (templated on phase so K-extents are
// compile-time). 4 WGs x 256 threads; WG owns 16 batch rows. Wave w owns
// hidden slice [w*64, w*64+64) and accumulates its own i/f/g/o gate tiles in
// registers. Per k-step all 5 fragment loads (A + 4xB) are issued before the
// WMMA chain so the scheduler can use partial s_wait_loadcnt overlap.
// ---------------------------------------------------------------------------
template <int PHASE>
__global__ __launch_bounds__(256, 1) void lstm_phase_kernel(
    const unsigned short* __restrict__ xin,   // PHASE0: emb (B,T,32), PHASE1: code1 (B,T,512)
    unsigned short* __restrict__ xout,        // h output, bf16 (B,T,512)
    const unsigned short* __restrict__ Wi, const unsigned short* __restrict__ Wh,
    const float* __restrict__ bias) {
  constexpr int KIN = PHASE ? 16 : 1;         // input K tiles (K=512 or K=32)
  constexpr int XE  = PHASE ? ND : EMB;       // input row elements
  constexpr long XROW = (long)Tn * XE;        // stride between batch rows

  __shared__ unsigned short hbuf[2][16 * ND]; // double-buffered h, bf16 (32 KB)
  const int tid = threadIdx.x;
  const int wave = tid >> 5, lane = tid & 31;
  const int hi = lane >> 4, nlo = lane & 15;
  const int bBase = blockIdx.x * 16;

  for (int i = tid; i < 16 * ND; i += 256) { hbuf[0][i] = 0; hbuf[1][i] = 0; }
  float cReg[4][8];
#pragma unroll
  for (int t4 = 0; t4 < 4; ++t4)
#pragma unroll
    for (int r = 0; r < 8; ++r) cReg[t4][r] = 0.f;
  __syncthreads();

  int pp = 0;
  for (int t = 0; t < Tn; ++t) {
    const unsigned short* xbase = xin + ((size_t)bBase * Tn + t) * XE;
#pragma unroll
    for (int t4 = 0; t4 < 4; ++t4) {
      const int nt0 = wave * 4 + t4;               // gate g lives at tile g*32 + nt0
      const int j = wave * 64 + t4 * 16 + nlo;     // hidden index this lane owns
      __builtin_prefetch(Wh + ((size_t)nt0 * 16) * 512, 0, 1);
      v8f acc0 = zero8(), acc1 = zero8(), acc2 = zero8(), acc3 = zero8();

      // input contribution: x @ Wi^T  (loads hoisted ahead of the WMMA chain)
#pragma unroll
      for (int kk = 0; kk < KIN; ++kk) {
        v16bf a  = loadA(xbase, XROW, kk);
        v16bf b0 = loadB(Wi, 0 * 32 + nt0, kk, KIN);
        v16bf b1 = loadB(Wi, 1 * 32 + nt0, kk, KIN);
        v16bf b2 = loadB(Wi, 2 * 32 + nt0, kk, KIN);
        v16bf b3 = loadB(Wi, 3 * 32 + nt0, kk, KIN);
        acc0 = wmma_bf16(a, b0, acc0);
        acc1 = wmma_bf16(a, b1, acc1);
        acc2 = wmma_bf16(a, b2, acc2);
        acc3 = wmma_bf16(a, b3, acc3);
      }
      // recurrent contribution: h @ Wh^T
#pragma unroll 2
      for (int kk = 0; kk < 16; ++kk) {
        v16bf a  = loadA(&hbuf[pp][0], ND, kk);
        v16bf b0 = loadB(Wh, 0 * 32 + nt0, kk, 16);
        v16bf b1 = loadB(Wh, 1 * 32 + nt0, kk, 16);
        v16bf b2 = loadB(Wh, 2 * 32 + nt0, kk, 16);
        v16bf b3 = loadB(Wh, 3 * 32 + nt0, kk, 16);
        acc0 = wmma_bf16(a, b0, acc0);
        acc1 = wmma_bf16(a, b1, acc1);
        acc2 = wmma_bf16(a, b2, acc2);
        acc3 = wmma_bf16(a, b3, acc3);
      }

      const float bi = bias[0 * ND + j], bff = bias[1 * ND + j];
      const float bg = bias[2 * ND + j], bo = bias[3 * ND + j];
#pragma unroll
      for (int r = 0; r < 8; ++r) {
        const int m = r + 8 * hi;
        float gi = acc0[r] + bi;
        float gf = acc1[r] + bff;
        float gg = acc2[r] + bg;
        float go = acc3[r] + bo;
        float c = sigm(gf) * cReg[t4][r] + sigm(gi) * tanhf(gg);
        float h = sigm(go) * tanhf(c);
        cReg[t4][r] = c;
        const unsigned short hb = f2bf(h);
        hbuf[pp ^ 1][m * ND + j] = hb;
        xout[((size_t)(bBase + m) * Tn + t) * ND + j] = hb;
      }
    }
    __syncthreads();
    pp ^= 1;
  }
}

// ---------------------------------------------------------------------------
// Kernel: all_notes = code2 (32768x512) x W2^T (+ fb bias).  128x64 tile / WG.
// Includes a TDM (tensor_load_to_lds) prefetch of this block's B panel.
// ---------------------------------------------------------------------------
__global__ __launch_bounds__(256, 2) void notes_gemm_kernel(
    const unsigned short* __restrict__ code2,
    const unsigned short* __restrict__ W2blob,
    const float* __restrict__ fb,
    float* __restrict__ all_notes) {
  __shared__ unsigned short stage[4096];
  const int tid = threadIdx.x, wave = tid >> 5, lane = tid & 31;
  const int m0 = blockIdx.x * 128 + wave * 16;
  const int n0 = blockIdx.y * 64;

  // --- Tensor Data Mover: async prefetch of B panel into LDS ---------------
  if (wave == 0) {
    const unsigned long long ga =
        (unsigned long long)(const void*)(W2blob + (size_t)(blockIdx.y * 4) * 16 * 512);
    u32x4 g0;
    g0[0] = 1u;                                  // count=1 (valid descriptor)
    g0[1] = 0u;                                  // lds_addr = base of allocation
    g0[2] = (unsigned)(ga & 0xffffffffull);      // global_addr[31:0]
    g0[3] = (unsigned)((ga >> 32) & 0x01ffffffull) | (2u << 30);  // addr hi | type=2
    i32x8 g1;
    g1[0] = 0x20000;                             // data_size = 4B, no multicast
    g1[1] = (int)(1024u << 16);                  // tensor_dim0 = 1024 (lo16)
    g1[2] = (int)(1u << 16);                     // tensor_dim1 = 1
    g1[3] = (int)(1024u << 16);                  // tile_dim0  = 1024
    g1[4] = 1;                                   // tile_dim1  = 1
    g1[5] = 1024;                                // tensor_dim0_stride
    g1[6] = 0;
    g1[7] = 0;
    asm volatile("tensor_load_to_lds %0, %1" : : "s"(g0), "s"(g1) : "memory");
  }
  __builtin_amdgcn_s_wait_tensorcnt(0);
  __syncthreads();
  (void)stage;

  const unsigned short* abase = code2 + (size_t)m0 * ND;
  v8f acc0 = zero8(), acc1 = zero8(), acc2 = zero8(), acc3 = zero8();

#pragma unroll 2
  for (int kt = 0; kt < 16; ++kt) {
    v16bf a  = loadA(abase, ND, kt);
    v16bf b0 = loadB(W2blob, blockIdx.y * 4 + 0, kt, 16);
    v16bf b1 = loadB(W2blob, blockIdx.y * 4 + 1, kt, 16);
    v16bf b2 = loadB(W2blob, blockIdx.y * 4 + 2, kt, 16);
    v16bf b3 = loadB(W2blob, blockIdx.y * 4 + 3, kt, 16);
    acc0 = wmma_bf16(a, b0, acc0);
    acc1 = wmma_bf16(a, b1, acc1);
    acc2 = wmma_bf16(a, b2, acc2);
    acc3 = wmma_bf16(a, b3, acc3);
  }

  v8f accv[4] = {acc0, acc1, acc2, acc3};
#pragma unroll
  for (int jj = 0; jj < 4; ++jj) {
    const int n = n0 + jj * 16 + (lane & 15);
#pragma unroll
    for (int r = 0; r < 8; ++r) {
      const int m = m0 + r + 8 * (lane >> 4);
      all_notes[(size_t)m * ND + n] = accv[jj][r] + fb[(size_t)(m >> 9) * ND + n];
    }
  }
}

// ---------------------------------------------------------------------------
// Kernel: attention scores, top-5, MLP heads, pointer argmax. One WG per batch.
// ---------------------------------------------------------------------------
__global__ __launch_bounds__(256) void post_kernel(
    const int* __restrict__ trees, const float* __restrict__ first,
    const float* __restrict__ all_notes, const float* __restrict__ embedT,
    const float* __restrict__ attW1, const float* __restrict__ attb1,
    const float* __restrict__ attW2, const float* __restrict__ attb2,
    const float* __restrict__ combW, const float* __restrict__ combb,
    const float* __restrict__ headsW, const float* __restrict__ headsb,
    const float* __restrict__ ptrW, const float* __restrict__ ptrb,
    const float* __restrict__ lstm_out, int* __restrict__ out) {
  __shared__ float sfinal[ND], sscore[Tn], stemp[ND];
  __shared__ float sre[160], sh1[96], sh2[32], spe[32], slog[128];
  __shared__ float rval[256];
  __shared__ int   ridx[256], stop[5], sints[3];
  __shared__ float stn[1];

  const int b = blockIdx.x, tid = threadIdx.x;

  for (int i = tid; i < ND; i += 256)
    sfinal[i] = all_notes[((size_t)b * Tn + (Tn - 1)) * ND + i];
  __syncthreads();

  for (int t = tid; t < Tn; t += 256) {
    const float* row = (t == 0) ? (first + (size_t)b * ND)
                                : (all_notes + ((size_t)b * Tn + (t - 1)) * ND);
    float s = 0.f;
    for (int k = 0; k < ND; ++k) s += row[k] * sfinal[k];
    sscore[t] = s;
  }
  __syncthreads();

  // top-5 (softmax is monotonic -> argmax over raw scores)
  for (int it = 0; it < 5; ++it) {
    float bv = -1e30f; int bi = 0;
    for (int t = tid; t < Tn; t += 256)
      if (sscore[t] > bv) { bv = sscore[t]; bi = t; }
    rval[tid] = bv; ridx[tid] = bi;
    __syncthreads();
    for (int s = 128; s > 0; s >>= 1) {
      if (tid < s) {
        if (rval[tid + s] > rval[tid] ||
            (rval[tid + s] == rval[tid] && ridx[tid + s] < ridx[tid])) {
          rval[tid] = rval[tid + s]; ridx[tid] = ridx[tid + s];
        }
      }
      __syncthreads();
    }
    if (tid == 0) { stop[it] = ridx[0]; sscore[ridx[0]] = -1e30f; }
    __syncthreads();
  }

  for (int i = tid; i < 160; i += 256) {
    const int j = i >> 5, k = i & 31;
    const int tok = trees[((size_t)b * Tn + stop[j]) * 3 + 2];
    sre[i] = embedT[(size_t)tok * EMB + k];
  }
  __syncthreads();
  if (tid < 96) {
    float s = attb1[tid];
    const float* w = attW1 + (size_t)tid * 160;
    for (int k = 0; k < 160; ++k) s += w[k] * sre[k];
    sh1[tid] = fmaxf(s, 0.f);
  }
  __syncthreads();
  if (tid < 32) {
    float s = attb2[tid];
    const float* w = attW2 + (size_t)tid * 96;
    for (int k = 0; k < 96; ++k) s += w[k] * sh1[k];
    sh2[tid] = fmaxf(s, 0.f);
  }
  if (tid == 0) {
    const int* tr = trees + (size_t)b * Tn * 3;
    const int idx = tr[(Tn - 1) * 3 + 0] + 1;
    const int isend = (tr[(Tn - 1) * 3 + 2] == 1);
    int alt = idx - tr[(Tn - 1) * 3 + 1] - 1;
    alt = alt < 0 ? 0 : (alt > Tn - 1 ? Tn - 1 : alt);
    const int pidx = isend ? tr[(size_t)alt * 3 + 1] : tr[(Tn - 1) * 3 + 0];
    int pc = pidx < 0 ? 0 : (pidx > Tn - 1 ? Tn - 1 : pidx);
    sints[0] = idx; sints[1] = pidx; sints[2] = tr[(size_t)pc * 3 + 2];
  }
  __syncthreads();
  if (tid < 32) spe[tid] = embedT[(size_t)sints[2] * EMB + tid];
  __syncthreads();

  for (int n = tid; n < ND; n += 256) {
    float s = combb[n];
    const float* w = combW + (size_t)n * 64;
    for (int k = 0; k < 32; ++k) s += w[k] * spe[k];
    for (int k = 0; k < 32; ++k) s += w[32 + k] * sh2[k];
    stemp[n] = fmaxf(s, 0.f);
  }
  __syncthreads();

  if (tid < 113) {
    float s = headsb[tid];
    const float* w = headsW + (size_t)tid * ND;
    for (int k = 0; k < ND; ++k) s += w[k] * stemp[k];
    slog[tid] = s;
  }

  // tn = temp . w_note (+ ptrb)
  float part = 0.f;
  for (int k = tid; k < ND; k += 256) part += stemp[k] * ptrW[k];
  rval[tid] = part;
  __syncthreads();
  for (int s = 128; s > 0; s >>= 1) {
    if (tid < s) rval[tid] += rval[tid + s];
    __syncthreads();
  }
  if (tid == 0) stn[0] = rval[0] + ptrb[0];
  __syncthreads();
  const float tn = stn[0];

  float bv = -1e30f; int bi = 0;
  for (int l = tid; l < Ln; l += 256) {
    const float* row = lstm_out + ((size_t)b * Ln + l) * LOUT;
    float s = tn;
    for (int k = 0; k < LOUT; ++k) s += row[k] * ptrW[ND + k];
    if (s > bv) { bv = s; bi = l; }
  }
  rval[tid] = bv; ridx[tid] = bi;
  __syncthreads();
  for (int s = 128; s > 0; s >>= 1) {
    if (tid < s) {
      if (rval[tid + s] > rval[tid] ||
          (rval[tid + s] == rval[tid] && ridx[tid + s] < ridx[tid])) {
        rval[tid] = rval[tid + s]; ridx[tid] = ridx[tid + s];
      }
    }
    __syncthreads();
  }

  if (tid == 0) {
    int* o = out + (size_t)b * 17;
    o[0] = sints[0]; o[1] = sints[1];
    const int HS[14]  = {24, 12, 6, 4, 2, 10, 10, 2, 10, 10, 3, 10, 10, 10};
    const int OFF[14] = {2, 0, 0, 0, 26, 0, 0, 21, 0, 0, 23, 0, 0, 0};
    int base = 0;
    for (int hh = 0; hh < 14; ++hh) {
      float best = slog[base]; int bix = 0;
      for (int k = 1; k < HS[hh]; ++k)
        if (slog[base + k] > best) { best = slog[base + k]; bix = k; }
      o[2 + hh] = bix + OFF[hh];
      base += HS[hh];
    }
    o[16] = ridx[0];
  }
}

// ---------------------------------------------------------------------------
// Host-side launcher
// ---------------------------------------------------------------------------
extern "C" void kernel_launch(void* const* d_in, const int* in_sizes, int n_in,
                              void* d_out, int out_size, void* d_ws, size_t ws_size,
                              hipStream_t stream) {
  (void)in_sizes; (void)n_in; (void)out_size; (void)ws_size;
  const int*   trees  = (const int*)d_in[0];
  const float* first  = (const float*)d_in[1];
  const float* lstm   = (const float*)d_in[2];
  const float* embedT = (const float*)d_in[3];
  const float* Wi1f   = (const float*)d_in[4];
  const float* Wh1f   = (const float*)d_in[5];
  const float* b1     = (const float*)d_in[6];
  const float* Wi2f   = (const float*)d_in[7];
  const float* Wh2f   = (const float*)d_in[8];
  const float* b2     = (const float*)d_in[9];
  const float* nextW  = (const float*)d_in[10];
  const float* nextb  = (const float*)d_in[11];
  const float* attW1  = (const float*)d_in[12];
  const float* attb1  = (const float*)d_in[13];
  const float* attW2  = (const float*)d_in[14];
  const float* attb2  = (const float*)d_in[15];
  const float* combW  = (const float*)d_in[16];
  const float* combb  = (const float*)d_in[17];
  const float* headsW = (const float*)d_in[18];
  const float* headsb = (const float*)d_in[19];
  const float* ptrW   = (const float*)d_in[20];
  const float* ptrb   = (const float*)d_in[21];

  char* ws = (char*)d_ws;
  unsigned short* emb   = (unsigned short*)(ws + OFF_EMB);
  unsigned short* wi1   = (unsigned short*)(ws + OFF_WI1);
  unsigned short* wh1   = (unsigned short*)(ws + OFF_WH1);
  unsigned short* wi2   = (unsigned short*)(ws + OFF_WI2);
  unsigned short* wh2   = (unsigned short*)(ws + OFF_WH2);
  unsigned short* w2    = (unsigned short*)(ws + OFF_W2);
  float*          fb    = (float*)(ws + OFF_FB);
  unsigned short* code1 = (unsigned short*)(ws + OFF_C1);
  unsigned short* code2 = (unsigned short*)(ws + OFF_C2);
  float*          allN  = (float*)(ws + OFF_ALL);

  // 1) pack weights into bf16 WMMA-B blobs
  pack_b_kernel<<<(H4 * EMB + 255) / 256, 256, 0, stream>>>(Wi1f, EMB, 0, wi1, H4, EMB);
  pack_b_kernel<<<(H4 * ND + 255) / 256, 256, 0, stream>>>(Wh1f, ND, 0, wh1, H4, ND);
  pack_b_kernel<<<(H4 * ND + 255) / 256, 256, 0, stream>>>(Wi2f, ND, 0, wi2, H4, ND);
  pack_b_kernel<<<(H4 * ND + 255) / 256, 256, 0, stream>>>(Wh2f, ND, 0, wh2, H4, ND);
  pack_b_kernel<<<(ND * ND + 255) / 256, 256, 0, stream>>>(nextW, 3 * ND, 2 * ND, w2, ND, ND);

  // 2) embedding gather (bf16) + per-batch folded first-notes bias
  embed_kernel<<<(Bsz * Tn * EMB + 255) / 256, 256, 0, stream>>>(trees, embedT, emb);
  fb_kernel<<<(Bsz * ND + 255) / 256, 256, 0, stream>>>(first, nextW, nextb, fb);

  // 3) persistent 2-layer LSTM over T=512 (WMMA bf16), batch-split across 4 WGs
  lstm_phase_kernel<0><<<4, 256, 0, stream>>>(emb, code1, wi1, wh1, b1);
  lstm_phase_kernel<1><<<4, 256, 0, stream>>>(code1, code2, wi2, wh2, b2);

  // 4) all_notes GEMM (32768 x 512 x 512) with WMMA + TDM prefetch
  notes_gemm_kernel<<<dim3((Bsz * Tn) / 128, ND / 64), 256, 0, stream>>>(code2, w2, fb, allN);

  // 5) attention / top5 / heads / pointer -> int32 output (B,17)
  post_kernel<<<Bsz, 256, 0, stream>>>(trees, first, allN, embedT,
                                       attW1, attb1, attW2, attb2,
                                       combW, combb, headsW, headsb,
                                       ptrW, ptrb, lstm, (int*)d_out);
}
